// FlowODEv2_27676769255885
// MI455X (gfx1250) — matrix operand order repaired and checked by reading, not compiled, not run on hardware
//
#include <hip/hip_runtime.h>
#include <hip/hip_bf16.h>
#include <math.h>

// ---- CDNA5 (gfx1250) wave32 WMMA types ----
typedef __attribute__((ext_vector_type(16))) __bf16 v16bf;
typedef __attribute__((ext_vector_type(8)))  __bf16 v8bf;
typedef __attribute__((ext_vector_type(4)))  __bf16 v4bf;
typedef __attribute__((ext_vector_type(8)))  float  v8f;

// ---- TDM descriptor vector types (clang-23 6-arg builtin form) ----
typedef unsigned int tdm_u32x4 __attribute__((ext_vector_type(4)));
typedef int          tdm_i32x8 __attribute__((ext_vector_type(8)));
typedef int          tdm_i32x4 __attribute__((ext_vector_type(4)));

#define BM 128
#define BN 128
#define BK 32
#define SSTR 40   // LDS row stride in bf16 elements (BK + 8 pad = 80B rows)

enum { EPI_STORE = 0, EPI_BIAS_GELU = 1, EPI_RESID = 2, EPI_ADDIN = 3 };

// ---------------------------------------------------------------------------
// f32-input WMMA GEMM: C(M,N) = epi( A(M,K) * B(N,K)^T ), bf16 convert on the
// fly, double-buffered LDS pipeline (loads for kt+1 issue before WMMAs of kt).
// ---------------------------------------------------------------------------
template <int EPI>
__global__ __launch_bounds__(256) void gemm_wmma_kernel(
    const float* __restrict__ A, int lda,
    const float* __restrict__ B, int ldb,
    float* __restrict__ C, int ldc,
    const float* __restrict__ Cin,
    const float* __restrict__ bias,
    const float* __restrict__ scale_logtau,
    int K)
{
  __shared__ __bf16 sA[2][BM * SSTR];
  __shared__ __bf16 sB[2][BN * SSTR];

  const int tid  = threadIdx.x;
  const int lane = tid & 31;
  const int wave = tid >> 5;
  const int g    = lane >> 4;
  const int lr   = lane & 15;
  const int wm   = wave & 3;    // M offset = wm*32
  const int wn   = wave >> 2;   // N offset = wn*64

  const int m0 = blockIdx.y * BM;
  const int n0 = blockIdx.x * BN;

  const int tcol = tid & 7;
  const int trow = tid >> 3;

  const float* aptr = A + (size_t)(m0 + trow) * lda + tcol * 4;
  const float* bptr = B + (size_t)(n0 + trow) * ldb + tcol * 4;

  v8f acc[2][4] = {};
  float4 ra[4], rb[4];

#pragma unroll
  for (int p = 0; p < 4; ++p) {
    ra[p] = *(const float4*)(aptr + (size_t)p * 32 * lda);
    rb[p] = *(const float4*)(bptr + (size_t)p * 32 * ldb);
  }
#pragma unroll
  for (int p = 0; p < 4; ++p) {
    v4bf ab = {(__bf16)ra[p].x, (__bf16)ra[p].y, (__bf16)ra[p].z, (__bf16)ra[p].w};
    *(v4bf*)(&sA[0][(trow + p * 32) * SSTR + tcol * 4]) = ab;
    v4bf bb = {(__bf16)rb[p].x, (__bf16)rb[p].y, (__bf16)rb[p].z, (__bf16)rb[p].w};
    *(v4bf*)(&sB[0][(trow + p * 32) * SSTR + tcol * 4]) = bb;
  }
  __syncthreads();

  const int KT = K / BK;
  for (int kt = 0; kt < KT; ++kt) {
    const int cur = kt & 1;

    if (kt + 1 < KT) {
      const int kn = (kt + 1) * BK;
#pragma unroll
      for (int p = 0; p < 4; ++p) {
        ra[p] = *(const float4*)(aptr + kn + (size_t)p * 32 * lda);
        rb[p] = *(const float4*)(bptr + kn + (size_t)p * 32 * ldb);
      }
      if (kt + 2 < KT) {
        __builtin_prefetch(aptr + (size_t)(kt + 2) * BK, 0, 1);
        __builtin_prefetch(bptr + (size_t)(kt + 2) * BK, 0, 1);
      }
    }

    v16bf af[2];
#pragma unroll
    for (int i = 0; i < 2; ++i) {
      const __bf16* pa = &sA[cur][(wm * 32 + i * 16 + lr) * SSTR + g * 8];
      v8bf lo = *(const v8bf*)pa;
      v8bf hi = *(const v8bf*)(pa + 16);
      af[i] = __builtin_shufflevector(lo, hi, 0, 1, 2, 3, 4, 5, 6, 7,
                                      8, 9, 10, 11, 12, 13, 14, 15);
    }
    v16bf bfr[4];
#pragma unroll
    for (int j = 0; j < 4; ++j) {
      const __bf16* pb = &sB[cur][(wn * 64 + j * 16 + lr) * SSTR + g * 8];
      v8bf lo = *(const v8bf*)pb;
      v8bf hi = *(const v8bf*)(pb + 16);
      bfr[j] = __builtin_shufflevector(lo, hi, 0, 1, 2, 3, 4, 5, 6, 7,
                                       8, 9, 10, 11, 12, 13, 14, 15);
    }

#pragma unroll
    for (int i = 0; i < 2; ++i)
#pragma unroll
      for (int j = 0; j < 4; ++j)
        acc[i][j] = __builtin_amdgcn_wmma_f32_16x16x32_bf16(
            false, af[i], false, bfr[j], (short)0, acc[i][j], false, false);

    if (kt + 1 < KT) {
      const int nxt = cur ^ 1;
#pragma unroll
      for (int p = 0; p < 4; ++p) {
        v4bf ab = {(__bf16)ra[p].x, (__bf16)ra[p].y, (__bf16)ra[p].z, (__bf16)ra[p].w};
        *(v4bf*)(&sA[nxt][(trow + p * 32) * SSTR + tcol * 4]) = ab;
        v4bf bb = {(__bf16)rb[p].x, (__bf16)rb[p].y, (__bf16)rb[p].z, (__bf16)rb[p].w};
        *(v4bf*)(&sB[nxt][(trow + p * 32) * SSTR + tcol * 4]) = bb;
      }
    }
    __syncthreads();
  }

  float scale = 0.0f;
  if (EPI == EPI_RESID) scale = 0.01f * expf(scale_logtau[0]);

#pragma unroll
  for (int i = 0; i < 2; ++i) {
#pragma unroll
    for (int j = 0; j < 4; ++j) {
#pragma unroll
      for (int r = 0; r < 8; ++r) {
        const int m = m0 + wm * 32 + i * 16 + g * 8 + r;
        const int n = n0 + wn * 64 + j * 16 + lr;
        float v = acc[i][j][r];
        if (EPI == EPI_BIAS_GELU) {
          v += bias[n];
          v = 0.5f * v * (1.0f + erff(v * 0.70710678118654752440f));
        } else if (EPI == EPI_RESID) {
          v = Cin[(size_t)m * ldc + n] + scale * (v + bias[n]);
        } else if (EPI == EPI_ADDIN) {
          v = Cin[(size_t)m * ldc + n] + v;
        }
        C[(size_t)m * ldc + n] = v;
      }
    }
  }
}

// ---------------------------------------------------------------------------
// TDM helper: DMA one 128x32 bf16 tile (row stride = stride_elems) into LDS
// at byte offset lds_addr, with hardware padding 16B per 64B row -> SSTR=40.
// D# layout per CDNA5 ISA ch.8.3/8.4 (group0 128b, group1 256b; groups 2/3
// zero => 2D tensor).
// ---------------------------------------------------------------------------
__device__ __forceinline__ void tdm_load_tile_bf16(
    const __bf16* gptr, unsigned lds_addr, int stride_elems)
{
  const unsigned long long ga = (unsigned long long)(uintptr_t)gptr;
  tdm_u32x4 g0;
  g0[0] = 1u;                          // count=1 (valid user descriptor)
  g0[1] = lds_addr;                    // lds_addr [63:32]
  g0[2] = (unsigned)ga;                // global_addr[31:0]
  g0[3] = (unsigned)((ga >> 32) & 0x01FFFFFFu) | (2u << 30);  // addr[56:32], type=2
  tdm_i32x8 g1;
  g1[0] = (int)((1u << 16)   |         // data_size = 2 bytes
                (1u << 20)   |         // pad_enable
                (3u << 22)   |         // pad_interval: 16 DWORDs (64B)
                (3u << 25));           // pad_amount:   4 DWORDs (16B)
  g1[1] = (int)(((unsigned)BK & 0xFFFFu) << 16);   // tensor_dim0[15:0]=32
  g1[2] = (int)(((unsigned)BM & 0xFFFFu) << 16);   // dim0 hi=0, tensor_dim1 lo=128
  g1[3] = (int)((unsigned)BK << 16);               // dim1 hi=0, tile_dim0=32
  g1[4] = (int)(unsigned)BM;                       // tile_dim1=128, tile_dim2=0
  g1[5] = (int)(unsigned)stride_elems;             // tensor_dim0_stride lo
  g1[6] = 0;                                       // stride hi / dim1_stride lo
  g1[7] = 0;
  tdm_i32x4 z4 = {0, 0, 0, 0};
  tdm_i32x8 z8 = {0, 0, 0, 0, 0, 0, 0, 0};
  // clang-23 / therock-10.0 form: (g0, g1, g2, g3, g4, cpol)
  __builtin_amdgcn_tensor_load_to_lds(g0, g1, z4, z4, z8, 0);
}

__device__ __forceinline__ unsigned lds_byte_off(const void* p) {
  // generic pointer to LDS: low 32 bits are the LDS byte offset (ISA 10.2)
  return (unsigned)(uintptr_t)p;
}

// ---------------------------------------------------------------------------
// bf16-input WMMA GEMM with Tensor-Data-Mover staging (dominant logits GEMM).
// A (M,K) bf16, B (N,K) bf16, C f32. Wave 0 drives the DMA; TENSORcnt +
// workgroup barrier publish each double-buffered tile.
// ---------------------------------------------------------------------------
__global__ __launch_bounds__(256) void gemm_wmma_bf16_tdm_kernel(
    const __bf16* __restrict__ A, int lda,
    const __bf16* __restrict__ B, int ldb,
    float* __restrict__ C, int ldc, int K)
{
  __shared__ __bf16 sA[2][BM * SSTR];
  __shared__ __bf16 sB[2][BN * SSTR];

  const int tid  = threadIdx.x;
  const int lane = tid & 31;
  const int wave = tid >> 5;
  const int g    = lane >> 4;
  const int lr   = lane & 15;
  const int wm   = wave & 3;
  const int wn   = wave >> 2;

  const int m0 = blockIdx.y * BM;
  const int n0 = blockIdx.x * BN;

  const __bf16* atile = A + (size_t)m0 * lda;
  const __bf16* btile = B + (size_t)n0 * ldb;

  v8f acc[2][4] = {};

  if (wave == 0) {
    tdm_load_tile_bf16(atile, lds_byte_off(&sA[0][0]), lda);
    tdm_load_tile_bf16(btile, lds_byte_off(&sB[0][0]), ldb);
    __builtin_amdgcn_s_wait_tensorcnt(0);
  }
  __syncthreads();

  const int KT = K / BK;
  for (int kt = 0; kt < KT; ++kt) {
    const int cur = kt & 1;

    // Kick off DMA of the next K-tile into the other buffer (wave 0 only).
    if (wave == 0 && kt + 1 < KT) {
      const int nxt = cur ^ 1;
      tdm_load_tile_bf16(atile + (size_t)(kt + 1) * BK,
                         lds_byte_off(&sA[nxt][0]), lda);
      tdm_load_tile_bf16(btile + (size_t)(kt + 1) * BK,
                         lds_byte_off(&sB[nxt][0]), ldb);
    }

    v16bf af[2];
#pragma unroll
    for (int i = 0; i < 2; ++i) {
      const __bf16* pa = &sA[cur][(wm * 32 + i * 16 + lr) * SSTR + g * 8];
      v8bf lo = *(const v8bf*)pa;
      v8bf hi = *(const v8bf*)(pa + 16);
      af[i] = __builtin_shufflevector(lo, hi, 0, 1, 2, 3, 4, 5, 6, 7,
                                      8, 9, 10, 11, 12, 13, 14, 15);
    }
    v16bf bfr[4];
#pragma unroll
    for (int j = 0; j < 4; ++j) {
      const __bf16* pb = &sB[cur][(wn * 64 + j * 16 + lr) * SSTR + g * 8];
      v8bf lo = *(const v8bf*)pb;
      v8bf hi = *(const v8bf*)(pb + 16);
      bfr[j] = __builtin_shufflevector(lo, hi, 0, 1, 2, 3, 4, 5, 6, 7,
                                       8, 9, 10, 11, 12, 13, 14, 15);
    }

#pragma unroll
    for (int i = 0; i < 2; ++i)
#pragma unroll
      for (int j = 0; j < 4; ++j)
        acc[i][j] = __builtin_amdgcn_wmma_f32_16x16x32_bf16(
            false, af[i], false, bfr[j], (short)0, acc[i][j], false, false);

    // DMA of next tile must have landed before the barrier publishes it.
    if (wave == 0) __builtin_amdgcn_s_wait_tensorcnt(0);
    __syncthreads();
  }

#pragma unroll
  for (int i = 0; i < 2; ++i)
#pragma unroll
    for (int j = 0; j < 4; ++j)
#pragma unroll
      for (int r = 0; r < 8; ++r) {
        const int m = m0 + wm * 32 + i * 16 + g * 8 + r;
        const int n = n0 + wn * 64 + j * 16 + lr;
        C[(size_t)m * ldc + n] = acc[i][j][r];
      }
}

// ---------------------------------------------------------------------------
// small kernels
// ---------------------------------------------------------------------------
__global__ __launch_bounds__(256) void gather_kernel(
    const int* __restrict__ tokens, const float* __restrict__ embed_w,
    float* __restrict__ x)
{
  const int idx = blockIdx.x * 256 + threadIdx.x;
  const int row = idx >> 9;
  const int c4  = idx & 511;
  const int tok = tokens[row];
  *(float4*)(x + (size_t)row * 2048 + c4 * 4) =
      *(const float4*)(embed_w + (size_t)tok * 2048 + c4 * 4);
}

__global__ __launch_bounds__(256) void bias_step_kernel(
    const float* __restrict__ W1, const float* __restrict__ b1,
    float* __restrict__ b1step, int step, int nsteps)
{
  const int h = blockIdx.x * 256 + threadIdx.x;
  const float t = (float)step / (float)(nsteps - 1);
  const float* w = W1 + (size_t)h * 320 + 256;
  float acc = b1[h];
#pragma unroll
  for (int j = 0; j < 32; ++j) {
    const float ang = t * (float)j * 6.283185307179586f;
    acc += w[j] * sinf(ang) + w[32 + j] * cosf(ang);
  }
  b1step[h] = acc;
}

__global__ __launch_bounds__(256) void rmsnorm_kernel(
    float* __restrict__ x, const float* __restrict__ nw)
{
  const int m = blockIdx.x;
  const int tid = threadIdx.x;
  float* row = x + (size_t)m * 2048;
  float4 a = *(float4*)(row + tid * 8);
  float4 b = *(float4*)(row + tid * 8 + 4);
  float ss = a.x * a.x + a.y * a.y + a.z * a.z + a.w * a.w +
             b.x * b.x + b.y * b.y + b.z * b.z + b.w * b.w;
  __shared__ float red[256];
  red[tid] = ss;
  __syncthreads();
  for (int s = 128; s > 0; s >>= 1) {
    if (tid < s) red[tid] += red[tid + s];
    __syncthreads();
  }
  const float r = rsqrtf(red[0] * (1.0f / 2048.0f) + 1.1920928955078125e-07f);
  const float4 wa = *(const float4*)(nw + tid * 8);
  const float4 wb = *(const float4*)(nw + tid * 8 + 4);
  a.x *= r * wa.x; a.y *= r * wa.y; a.z *= r * wa.z; a.w *= r * wa.w;
  b.x *= r * wb.x; b.y *= r * wb.y; b.z *= r * wb.z; b.w *= r * wb.w;
  *(float4*)(row + tid * 8) = a;
  *(float4*)(row + tid * 8 + 4) = b;
}

// 8 elements per thread f32 -> bf16
__global__ __launch_bounds__(256) void f32_to_bf16_kernel(
    const float* __restrict__ src, __bf16* __restrict__ dst)
{
  const size_t i = ((size_t)blockIdx.x * 256 + threadIdx.x) * 8;
  const float4 a = *(const float4*)(src + i);
  const float4 b = *(const float4*)(src + i + 4);
  v4bf pa = {(__bf16)a.x, (__bf16)a.y, (__bf16)a.z, (__bf16)a.w};
  v4bf pb = {(__bf16)b.x, (__bf16)b.y, (__bf16)b.z, (__bf16)b.w};
  *(v4bf*)(dst + i) = pa;
  *(v4bf*)(dst + i + 4) = pb;
}

extern "C" void kernel_launch(void* const* d_in, const int* in_sizes, int n_in,
                              void* d_out, int out_size, void* d_ws, size_t ws_size,
                              hipStream_t stream) {
  (void)in_sizes; (void)n_in; (void)out_size;
  const int*   tokens    = (const int*)d_in[0];
  const float* embed_w   = (const float*)d_in[1];
  const float* lm_head_w = (const float*)d_in[2];
  const float* norm_w    = (const float*)d_in[3];
  const float* P         = (const float*)d_in[4];
  const float* Q         = (const float*)d_in[5];
  const float* W1        = (const float*)d_in[6];
  const float* b1        = (const float*)d_in[7];
  const float* W2        = (const float*)d_in[8];
  const float* b2        = (const float*)d_in[9];
  const float* log_tau   = (const float*)d_in[10];
  float* logits = (float*)d_out;

  const int BT = 4096, D = 2048, R = 256, H = 1024, V = 32000, NSTEPS = 16;

  char* ws = (char*)d_ws;
  const size_t xB = (size_t)BT * D * 4, zB = (size_t)BT * R * 4,
               hB = (size_t)BT * H * 4, bB = 4096;
  float* x      = (float*)(ws);
  float* z      = (float*)(ws + xB);
  float* h      = (float*)(ws + xB + zB);
  float* b1step = (float*)(ws + xB + zB + hB);
  const size_t f32End = xB + zB + hB + bB;
  __bf16* x16  = (__bf16*)(ws + f32End);
  __bf16* lm16 = (__bf16*)(ws + f32End + (size_t)BT * D * 2);
  const size_t need = f32End + (size_t)BT * D * 2 + (size_t)V * D * 2;
  const bool use_tdm = (ws_size >= need);

  // 1) embedding gather
  gather_kernel<<<dim3((BT * D / 4) / 256), 256, 0, stream>>>(tokens, embed_w, x);

  // (L2-resident bf16 weight copy: 131MB < 192MB L2)
  if (use_tdm)
    f32_to_bf16_kernel<<<dim3((V * D) / 2048), 256, 0, stream>>>(lm_head_w, lm16);

  // 2) z0 = x @ P^T
  gemm_wmma_kernel<EPI_STORE><<<dim3(R / BN, BT / BM), 256, 0, stream>>>(
      x, D, P, D, z, R, nullptr, nullptr, nullptr, D);

  // 3) 16 ODE steps
  for (int s = 0; s < NSTEPS; ++s) {
    bias_step_kernel<<<dim3(H / 256), 256, 0, stream>>>(W1, b1, b1step, s, NSTEPS);
    gemm_wmma_kernel<EPI_BIAS_GELU><<<dim3(H / BN, BT / BM), 256, 0, stream>>>(
        z, R, W1, R + 64, h, H, nullptr, b1step, nullptr, R);
    gemm_wmma_kernel<EPI_RESID><<<dim3(R / BN, BT / BM), 256, 0, stream>>>(
        h, H, W2, H, z, R, z, b2, log_tau + s, H);
  }

  // 4) x = x + z @ Q^T (in place), 5) RMSNorm in place
  gemm_wmma_kernel<EPI_ADDIN><<<dim3(D / BN, BT / BM), 256, 0, stream>>>(
      z, R, Q, R, x, D, x, nullptr, nullptr, R);
  rmsnorm_kernel<<<dim3(BT), 256, 0, stream>>>(x, norm_w);

  // 6) logits = normed @ lm_head^T — dominant GEMM
  if (use_tdm) {
    f32_to_bf16_kernel<<<dim3((BT * D) / 2048), 256, 0, stream>>>(x, x16);
    gemm_wmma_bf16_tdm_kernel<<<dim3(V / BN, BT / BM), 256, 0, stream>>>(
        x16, D, lm16, D, logits, V, D);
  } else {
    gemm_wmma_kernel<EPI_STORE><<<dim3(V / BN, BT / BM), 256, 0, stream>>>(
        x, D, lm_head_w, D, logits, V, nullptr, nullptr, nullptr, D);
  }
}
